// LSTM_91027536871504
// MI455X (gfx1250) — matrix operand order repaired and checked by reading, not compiled, not run on hardware
//
#include <hip/hip_runtime.h>
#include <hip/hip_bf16.h>

typedef __attribute__((ext_vector_type(16))) _Float16 v16h;
typedef __attribute__((ext_vector_type(8)))  _Float16 v8h;
typedef __attribute__((ext_vector_type(4)))  _Float16 v4h;
typedef __attribute__((ext_vector_type(8)))  float    v8f;
typedef __attribute__((ext_vector_type(4)))  float    v4f;
typedef __attribute__((ext_vector_type(4)))  int      v4i;

typedef __attribute__((address_space(1))) v4i* gptr_v4i;  // global (prints as __device__)
typedef __attribute__((address_space(3))) v4i* lptr_v4i;  // LDS

#define Bdim 32768
#define Kdim 256     // I == H == 256 (K of each GEMM)
#define Hdim 256

#if __has_builtin(__builtin_amdgcn_global_load_async_to_lds_b128)
#define HAVE_ASYNC_LDS 1
#else
#define HAVE_ASYNC_LDS 0
#endif

__device__ __forceinline__ void async_wait0() {
#if __has_builtin(__builtin_amdgcn_s_wait_asynccnt)
  __builtin_amdgcn_s_wait_asynccnt(0);
#elif HAVE_ASYNC_LDS
  asm volatile("s_wait_asynccnt 0x0" ::: "memory");
#endif
}

// ---------------------------------------------------------------------------
// Prep: WT[n*K + k] = (f16) W[k*N + n]   (transpose + fp32->f16 convert)
// ---------------------------------------------------------------------------
__global__ void __launch_bounds__(256) transpose_cvt_f16(
    const float* __restrict__ W, _Float16* __restrict__ WT, int K, int N) {
  int idx = blockIdx.x * 256 + threadIdx.x;
  if (idx >= K * N) return;
  int n = idx / K;
  int k = idx - n * K;
  WT[idx] = (_Float16)W[(size_t)k * N + n];
}

// ---------------------------------------------------------------------------
// Fragment loaders from LDS tiles of row-stride 32 halves (64 B).
// A (16x32 f16): lane<16 -> row=lane,     K {0..7, 16..23}
//                lane>=16 -> row=lane-16, K {8..15, 24..31}
// B (32x16 f16): lane<16 -> col=lane,     K {0..7, 16..23}
//                lane>=16 -> col=lane-16, K {8..15, 24..31}
// ---------------------------------------------------------------------------
__device__ __forceinline__ v16h frag_a(const _Float16* base, int row0, int lane) {
  const int r = row0 + (lane & 15);
  const _Float16* p = base + r * 32 + ((lane >= 16) ? 8 : 0);
  v8h lo = *(const v8h*)p;
  v8h hh = *(const v8h*)(p + 16);
  v16h out;
#pragma unroll
  for (int j = 0; j < 8; ++j) { out[j] = lo[j]; out[j + 8] = hh[j]; }
  return out;
}

__device__ __forceinline__ v16h frag_b(const _Float16* base, int row0, int lane) {
  const int r = row0 + (lane & 15);
  const _Float16* p = base + r * 32 + ((lane >= 16) ? 8 : 0);
  v8h lo = *(const v8h*)p;
  v8h hh = *(const v8h*)(p + 16);
  v16h out;
#pragma unroll
  for (int j = 0; j < 8; ++j) { out[j] = lo[j]; out[j + 8] = hh[j]; }
  return out;
}

__device__ __forceinline__ float fast_sigmoid(float v) {
  v = fminf(fmaxf(v, -30.f), 30.f);
  return 1.f / (1.f + __expf(-v));
}
__device__ __forceinline__ float fast_tanh(float v) {
  v = fminf(fmaxf(v, -15.f), 15.f);
  float e = __expf(-2.f * v);
  return (1.f - e) / (1.f + e);
}

// Per-buffer LDS layout (halves): XS[64][32] HS[64][32] CS[64][32]
//                                 WX[4*64][32] WH[4*64][32] WC[3*64][32] = 56 KB
#define OFF_XS 0
#define OFF_HS 2048
#define OFF_CS 4096
#define OFF_WX 6144
#define OFF_WH 14336
#define OFF_WC 22528
#define BUF_HALVES 28672

// ---- pipelined activation staging: issue global loads ... convert+commit ----
__device__ __forceinline__ void act_issue(const float* __restrict__ x,
                                          const float* __restrict__ h,
                                          const float* __restrict__ c_prev,
                                          int mBase, int k0, int tid, v4f regs[6]) {
#pragma unroll
  for (int it = 0; it < 6; ++it) {
    int t   = tid + it * 256;
    int mat = t >> 9;                 // 512 tasks per matrix
    int rem = t & 511;
    int r   = rem >> 3;
    int g   = rem & 7;
    const float* src = (mat == 0) ? x : (mat == 1) ? h : c_prev;
    regs[it] = *(const v4f*)(src + (size_t)(mBase + r) * Kdim + k0 + g * 4);
  }
}

__device__ __forceinline__ void act_commit(_Float16* buf, int tid, const v4f regs[6]) {
#pragma unroll
  for (int it = 0; it < 6; ++it) {
    int t   = tid + it * 256;
    int mat = t >> 9;
    int rem = t & 511;
    int r   = rem >> 3;
    int g   = rem & 7;
    _Float16* dst = buf + ((mat == 0) ? OFF_XS : (mat == 1) ? OFF_HS : OFF_CS);
    v4h hv;
#pragma unroll
    for (int j = 0; j < 4; ++j) hv[j] = (_Float16)regs[it][j];
    *(v4h*)(dst + r * 32 + g * 4) = hv;
  }
}

// ---- weight staging: async DMA global->LDS (f16, no conversion needed) ----
__device__ __forceinline__ void stage_weights(const _Float16* __restrict__ WxT,
                                              const _Float16* __restrict__ WhT,
                                              const _Float16* __restrict__ WcT,
                                              _Float16* buf, int nBase, int k0, int tid) {
#pragma unroll
  for (int it = 0; it < 11; ++it) {
    int t = tid + it * 256;
    const _Float16* src;
    int ldsOff;
    if (t < 1024) {
      int lr = t >> 2, q = t & 3;
      int gate = lr >> 6, n = lr & 63;
      src    = WxT + (size_t)(gate * Hdim + nBase + n) * Kdim + k0 + q * 8;
      ldsOff = OFF_WX + lr * 32 + q * 8;
    } else if (t < 2048) {
      int t2 = t - 1024;
      int lr = t2 >> 2, q = t2 & 3;
      int gate = lr >> 6, n = lr & 63;
      src    = WhT + (size_t)(gate * Hdim + nBase + n) * Kdim + k0 + q * 8;
      ldsOff = OFF_WH + lr * 32 + q * 8;
    } else {  // t < 2816 always (max 2815)
      int t2 = t - 2048;
      int lr = t2 >> 2, q = t2 & 3;
      int gate = lr >> 6, n = lr & 63;
      src    = WcT + (size_t)(gate * Hdim + nBase + n) * Kdim + k0 + q * 8;
      ldsOff = OFF_WC + lr * 32 + q * 8;
    }
#if HAVE_ASYNC_LDS
    __builtin_amdgcn_global_load_async_to_lds_b128(
        (gptr_v4i)(void*)src, (lptr_v4i)(void*)(buf + ldsOff), 0, 0);
#else
    *(v8h*)(buf + ldsOff) = *(const v8h*)src;
#endif
  }
}

// ---- per-wave fragments + 22 WMMAs for one K-tile ----
__device__ __forceinline__ void compute_tile(const _Float16* buf, int wm, int wn, int lane,
                                             v8f accz[4][2], v8f accp[3][2]) {
  v16h xa0 = frag_a(buf + OFF_XS, wm * 32, lane);
  v16h xa1 = frag_a(buf + OFF_XS, wm * 32 + 16, lane);
  v16h ha0 = frag_a(buf + OFF_HS, wm * 32, lane);
  v16h ha1 = frag_a(buf + OFF_HS, wm * 32 + 16, lane);
  v16h ca0 = frag_a(buf + OFF_CS, wm * 32, lane);
  v16h ca1 = frag_a(buf + OFF_CS, wm * 32 + 16, lane);

#pragma unroll
  for (int g = 0; g < 4; ++g) {
    v16h bwx = frag_b(buf + OFF_WX, g * 64 + wn * 16, lane);
    accz[g][0] = __builtin_amdgcn_wmma_f32_16x16x32_f16(false, xa0, false, bwx,
                                                        (short)0, accz[g][0], false, false);
    accz[g][1] = __builtin_amdgcn_wmma_f32_16x16x32_f16(false, xa1, false, bwx,
                                                        (short)0, accz[g][1], false, false);
    v16h bwh = frag_b(buf + OFF_WH, g * 64 + wn * 16, lane);
    accz[g][0] = __builtin_amdgcn_wmma_f32_16x16x32_f16(false, ha0, false, bwh,
                                                        (short)0, accz[g][0], false, false);
    accz[g][1] = __builtin_amdgcn_wmma_f32_16x16x32_f16(false, ha1, false, bwh,
                                                        (short)0, accz[g][1], false, false);
  }
#pragma unroll
  for (int g = 0; g < 3; ++g) {
    v16h bwc = frag_b(buf + OFF_WC, g * 64 + wn * 16, lane);
    accp[g][0] = __builtin_amdgcn_wmma_f32_16x16x32_f16(false, ca0, false, bwc,
                                                        (short)0, accp[g][0], false, false);
    accp[g][1] = __builtin_amdgcn_wmma_f32_16x16x32_f16(false, ca1, false, bwc,
                                                        (short)0, accp[g][1], false, false);
  }
}

__global__ void __launch_bounds__(256) lstm_wmma_f16(
    const float* __restrict__ x, const float* __restrict__ h,
    const float* __restrict__ c_prev, const float* __restrict__ bx,
    const _Float16* __restrict__ WxT, const _Float16* __restrict__ WhT,
    const _Float16* __restrict__ WcT, float* __restrict__ out) {
  __shared__ _Float16 smem[2][BUF_HALVES];   // 112 KB, double-buffered

  const int tid   = threadIdx.x;
  const int nBase = blockIdx.x * 64;   // gate-column tile (within H)
  const int mBase = blockIdx.y * 64;   // batch-row tile
  const int lane  = tid & 31;
  const int wv    = tid >> 5;          // 8 waves
  const int wm    = wv & 1;            // wave row   (2)
  const int wn    = wv >> 1;           // wave col   (4)

  v8f accz[4][2];   // gates i,f,g,o of z = x@Wx + h@Wh
  v8f accp[3][2];   // peephole i,f,o of p = c_prev@Wc
#pragma unroll
  for (int g = 0; g < 4; ++g)
#pragma unroll
    for (int m = 0; m < 2; ++m) accz[g][m] = (v8f)(0.f);
#pragma unroll
  for (int g = 0; g < 3; ++g)
#pragma unroll
    for (int m = 0; m < 2; ++m) accp[g][m] = (v8f)(0.f);

  v4f areg[6];

  // ---- prologue: stage K-tile 0 into buffer 0 ----
  stage_weights(WxT, WhT, WcT, &smem[0][0], nBase, 0, tid);
  act_issue(x, h, c_prev, mBase, 0, tid, areg);
  act_commit(&smem[0][0], tid, areg);
  async_wait0();
  __syncthreads();

  // ---- pipelined main loop: prefetch kt+1 while computing kt ----
  for (int kt = 0; kt < Kdim / 32; ++kt) {
    _Float16* buf  = &smem[kt & 1][0];
    _Float16* nbuf = &smem[(kt & 1) ^ 1][0];
    if (kt < Kdim / 32 - 1) {
      stage_weights(WxT, WhT, WcT, nbuf, nBase, (kt + 1) * 32, tid);  // async DMA
      act_issue(x, h, c_prev, mBase, (kt + 1) * 32, tid, areg);       // loads in flight
    }
    compute_tile(buf, wm, wn, lane, accz, accp);                      // 22 WMMAs
    if (kt < Kdim / 32 - 1) act_commit(nbuf, tid, areg);              // cvt + ds_store
    async_wait0();
    __syncthreads();
  }

  // ---- epilogue: C-layout row = r + (lane>=16 ? 8:0), col = lane%16 ----
  const int  col = nBase + wn * 16 + (lane & 15);
  const int  rhi = (lane >= 16) ? 8 : 0;
  const float bi = bx[col];
  const float bf = bx[Hdim + col];
  const float bg = bx[2 * Hdim + col];
  const float bo = bx[3 * Hdim + col];
  float* outH = out;
  float* outC = out + (size_t)Bdim * Hdim;

#pragma unroll
  for (int mt = 0; mt < 2; ++mt) {
#pragma unroll
    for (int r = 0; r < 8; ++r) {
      int row = mBase + wm * 32 + mt * 16 + r + rhi;
      size_t idx = (size_t)row * Hdim + col;
      float cp = c_prev[idx];
      float zi = accz[0][mt][r] + accp[0][mt][r] + bi;
      float zf = accz[1][mt][r] + accp[1][mt][r] + bf;
      float zg = accz[2][mt][r] + bg;
      float zo = accz[3][mt][r] + accp[2][mt][r] + bo;
      float ig = fast_sigmoid(zi);
      float fg = fast_sigmoid(zf);
      float gg = fast_tanh(zg);
      float og = fast_sigmoid(zo);
      float cn = fmaf(ig, gg, fg * cp);
      float hn = og * fast_tanh(cn);
      outH[idx] = hn;
      outC[idx] = cn;
    }
  }
}

extern "C" void kernel_launch(void* const* d_in, const int* in_sizes, int n_in,
                              void* d_out, int out_size, void* d_ws, size_t ws_size,
                              hipStream_t stream) {
  const float* x  = (const float*)d_in[0];
  const float* h  = (const float*)d_in[1];
  const float* c  = (const float*)d_in[2];
  const float* Wx = (const float*)d_in[3];
  const float* bx = (const float*)d_in[4];
  const float* Wh = (const float*)d_in[5];
  const float* Wc = (const float*)d_in[6];

  _Float16* wsWxT = (_Float16*)d_ws;                  // [4H][K] f16
  _Float16* wsWhT = wsWxT + (size_t)4 * Hdim * Kdim;  // [4H][K] f16
  _Float16* wsWcT = wsWhT + (size_t)4 * Hdim * Kdim;  // [3H][K] f16

  transpose_cvt_f16<<<(Kdim * 4 * Hdim + 255) / 256, 256, 0, stream>>>(Wx, wsWxT, Kdim, 4 * Hdim);
  transpose_cvt_f16<<<(Kdim * 4 * Hdim + 255) / 256, 256, 0, stream>>>(Wh, wsWhT, Kdim, 4 * Hdim);
  transpose_cvt_f16<<<(Kdim * 3 * Hdim + 255) / 256, 256, 0, stream>>>(Wc, wsWcT, Kdim, 3 * Hdim);

  dim3 grid(Hdim / 64, Bdim / 64);  // N-tiles fastest -> row-block L2 reuse
  lstm_wmma_f16<<<grid, 256, 0, stream>>>(x, h, c, bx, wsWxT, wsWhT, wsWcT, (float*)d_out);
}
